// HyperNetwork_4174708212439
// MI455X (gfx1250) — matrix operand order repaired and checked by reading, not compiled, not run on hardware
//
#include <hip/hip_runtime.h>
#include <hip/hip_bf16.h>
#include <stdint.h>

// ---------------------------------------------------------------------------
// HyperNetwork on MI455X (gfx1250, wave32, WMMA, async-to-LDS)
//
// Dominant cost: p = h1 @ W2^T  (8192x512 x 512x21504 = 180 GFLOP) -> bf16
// WMMA (v_wmma_f32_16x16x32_bf16, fp32 accum), fully fused with the
// per-sample low-rank target net so p (704 MB fp32) never touches HBM.
// W2 (22 MB bf16) is L2-resident (192 MB L2). Per workgroup: 4 waves x 16
// samples share each staged 16x512 W2 slice (B operand) -> 1/4 L2 traffic.
// Each wave keeps its 16x512 H1 A-tile resident in 128 VGPRs for the whole
// 1344-chunk sweep. W2 slices are DOUBLE-BUFFERED in LDS and staged with
// GLOBAL_LOAD_ASYNC_TO_LDS_B128 (ASYNCcnt) so staging chunk t+1 overlaps the
// 16-WMMA consume of chunk t; one s_wait_asynccnt + one barrier per chunk.
// ---------------------------------------------------------------------------

typedef __bf16 bf16_t;
typedef __attribute__((ext_vector_type(16))) __bf16 v16bf;
typedef __attribute__((ext_vector_type(8)))  float  v8f;
typedef int v4i_vs __attribute__((vector_size(16)));  // matches builtin proto

#define B_TOT   8192
#define IN_DIM  256
#define P_DIM   128
#define H_DIM   512
#define TOTP    21504
#define OUT_DIM 64
#define NCHUNK  1344   // 21504 / 16 column-chunks of p

// --- CDNA5 async-to-LDS path (guarded; falls back to plain LDS copy) -------
#if defined(__AMDGCN__) && \
    __has_builtin(__builtin_amdgcn_global_load_async_to_lds_b128) && \
    __has_builtin(__builtin_amdgcn_s_wait_asynccnt)
#define HAVE_ASYNC 1
#else
#define HAVE_ASYNC 0
#endif

#if HAVE_ASYNC
#define ASYNC_WAIT() __builtin_amdgcn_s_wait_asynccnt(0)
typedef __attribute__((address_space(1))) v4i_vs* gas_v4i;  // global (AS1)
typedef __attribute__((address_space(3))) v4i_vs* las_v4i;  // LDS (AS3)
#else
#define ASYNC_WAIT()
#endif

// ---- WMMA helpers ---------------------------------------------------------
// D = A(16x32 bf16) x B(32x16 bf16) + C(16x16 f32)
__device__ __forceinline__ v8f wmma_bf16(v16bf a, v16bf b, v8f c) {
  return __builtin_amdgcn_wmma_f32_16x16x32_bf16(
      /*neg_a=*/false, a, /*neg_b=*/false, b,
      /*c_mod=*/(short)0, c, /*reuse_a=*/false, /*reuse_b=*/false);
}

// A-operand (16x32, row = sample) per ISA 7.12.2 16-bit A layout:
// lane m=lane&15; lanes 0-15 hold K [kb*32+0..7] U [kb*32+16..23],
// lanes 16-31 hold K [+8..15] U [+24..31].  Source is row-major.
__device__ __forceinline__ v16bf load_a16(const bf16_t* base, int rowstride,
                                          int kb, int lane) {
  const int m  = lane & 15;
  const int hi = lane >> 4;
  const bf16_t* p = base + (size_t)m * rowstride + (kb << 5) + (hi << 3);
  union { uint4 u[2]; v16bf v; } t;
  t.u[0] = *(const uint4*)(p);
  t.u[1] = *(const uint4*)(p + 16);
  return t.v;
}

// B-operand (32x16, col n = lane&15): same K order as A
// (lanes 0-15 hold K 0..7 U 16..23, lanes 16-31 hold K 8..15 U 24..31).
__device__ __forceinline__ v16bf load_b16(const bf16_t* base, int rowstride,
                                          int kb, int lane) {
  const int n  = lane & 15;
  const int hi = lane >> 4;
  const bf16_t* p = base + (size_t)n * rowstride + (kb << 5) + (hi << 3);
  union { uint4 u[2]; v16bf v; } t;
  t.u[0] = *(const uint4*)(p);
  t.u[1] = *(const uint4*)(p + 16);
  return t.v;
}

// ---- fp32 -> bf16 conversion ---------------------------------------------
__global__ void cvt_f32_bf16(const float* __restrict__ s,
                             bf16_t* __restrict__ d, int n) {
  int i = blockIdx.x * blockDim.x + threadIdx.x;
  const int stride = gridDim.x * blockDim.x;
  for (; i < n; i += stride) d[i] = (bf16_t)s[i];
}

// ---- Hypernetwork layers 0/1: H1 = relu(relu(P W0^T + b0) W1^T + b1) ------
// 2 waves / block, 16 samples per wave.
__global__ __launch_bounds__(64) void hyper_kernel(
    const bf16_t* __restrict__ Pbf, const bf16_t* __restrict__ W0bf,
    const float* __restrict__ b0,   const bf16_t* __restrict__ W1bf,
    const float* __restrict__ b1,   bf16_t* __restrict__ H1bf) {
  __shared__ bf16_t h0buf[2][16][H_DIM];  // 32 KB

  const int tid  = threadIdx.x;
  const int w    = tid >> 5;
  const int lane = tid & 31;
  const int n    = lane & 15;
  const int hi   = lane >> 4;
  const int base = (blockIdx.x * 2 + w) * 16;

  // layer 0: A = parameterizer tile (16 x 128), K = 128
  v16bf a0[4];
#pragma unroll
  for (int kb = 0; kb < 4; ++kb)
    a0[kb] = load_a16(Pbf + (size_t)base * P_DIM, P_DIM, kb, lane);

  for (int c = 0; c < 32; ++c) {
    v8f pc = {0.f, 0.f, 0.f, 0.f, 0.f, 0.f, 0.f, 0.f};
#pragma unroll
    for (int kb = 0; kb < 4; ++kb)
      pc = wmma_bf16(a0[kb],
                     load_b16(W0bf + (size_t)(c * 16) * P_DIM, P_DIM, kb, lane),
                     pc);
    const float bias = b0[c * 16 + n];
#pragma unroll
    for (int i = 0; i < 8; ++i) {
      float v = fmaxf(pc[i] + bias, 0.f);
      h0buf[w][i + 8 * hi][c * 16 + n] = (bf16_t)v;
    }
  }

  // layer 1: A = h0 tile (16 x 512), K = 512
  v16bf a1[16];
#pragma unroll
  for (int kb = 0; kb < 16; ++kb)
    a1[kb] = load_a16(&h0buf[w][0][0], H_DIM, kb, lane);

  for (int c = 0; c < 32; ++c) {
    v8f pc = {0.f, 0.f, 0.f, 0.f, 0.f, 0.f, 0.f, 0.f};
#pragma unroll
    for (int kb = 0; kb < 16; ++kb)
      pc = wmma_bf16(a1[kb],
                     load_b16(W1bf + (size_t)(c * 16) * H_DIM, H_DIM, kb, lane),
                     pc);
    const float bias = b1[c * 16 + n];
#pragma unroll
    for (int i = 0; i < 8; ++i) {
      float v = fmaxf(pc[i] + bias, 0.f);
      H1bf[(size_t)(base + i + 8 * hi) * H_DIM + c * 16 + n] = (bf16_t)v;
    }
  }
}

// ---- flat chunk schedule over p's 1344 16-column chunks -------------------
// Per layer l (512/512/320 chunks): first 256 chunks sweep the A-factor
// (base = offA + d*16), then the B-factor in (j0 outer, r inner) order
// (base = offA + 4096 + r*d2 + j0*16).
__device__ __forceinline__ int chunk_base(int t) {
  const int l  = (t < 512) ? 0 : (t < 1024 ? 1 : 2);
  const int t0 = t - l * 512;
  const int oa = l * 8192;
  const int d2 = (l == 2) ? OUT_DIM : IN_DIM;
  if (t0 < 256) return oa + t0 * 16;
  const int u = t0 - 256;
  return oa + 4096 + (u & 15) * d2 + (u >> 4) * 16;
}

// ---- stage one 16x512 bf16 W2 slice (16 KB) into an LDS buffer ------------
// Async path: GLOBAL_LOAD_ASYNC_TO_LDS_B128, memory->LDS without VGPR
// round-trip, tracked by ASYNCcnt. AS pointers built via uintptr_t: generic
// LDS addresses carry the LDS byte offset in bits[31:0] (ISA 10.2 aperture);
// global generic addresses equal their AS1 addresses.
__device__ __forceinline__ void stage_chunk(const bf16_t* __restrict__ w2,
                                            int colbase,
                                            bf16_t* __restrict__ dst,
                                            int tid) {
  for (int i = tid; i < 1024; i += 128) {  // 1024 x 16B
    const bf16_t* g = w2 + (size_t)(colbase + (i >> 6)) * H_DIM +
                      ((i & 63) << 3);
    bf16_t* l = dst + (i << 3);
#if HAVE_ASYNC
    __builtin_amdgcn_global_load_async_to_lds_b128(
        (gas_v4i)(uintptr_t)g, (las_v4i)(uintptr_t)l, 0, 0);
#else
    *(uint4*)l = *(const uint4*)g;
#endif
  }
}

// ---- Fused p-GEMM + per-sample low-rank target net ------------------------
// 4 waves / block, 64 samples / block. Double-buffered W2 staging:
// iteration t stages chunk t+1 into buf[(t+1)&1], consumes buf[t&1] with 16
// WMMAs + VALU contraction, then s_wait_asynccnt + one barrier.
__global__ __launch_bounds__(128) void target_kernel(
    const float* __restrict__ obs, const bf16_t* __restrict__ H1bf,
    const bf16_t* __restrict__ W2bf, const float* __restrict__ b2,
    float* __restrict__ out) {
  __shared__ bf16_t xbuf[64][IN_DIM];      // 32 KB, per-wave 16-row slices
  __shared__ bf16_t w2s[2][16][H_DIM];     // 2 x 16 KB, shared B operand
  __shared__ float  ybuf[4][16][16];       //  4 KB, per-wave rank-16 activ.

  const int tid   = threadIdx.x;
  const int w     = tid >> 5;
  const int lane  = tid & 31;
  const int n     = lane & 15;   // WMMA D: lane -> output column
  const int hi    = lane >> 4;   // WMMA D: lane group -> rows 0-7 / 8-15
  const int gbase = blockIdx.x * 64;
  const int wbase = w * 16;

  // x <- obs tile, converted to bf16
  for (int i = tid; i < 64 * IN_DIM; i += 128) {
    const int r = i >> 8, c = i & 255;
    xbuf[r][c] = (bf16_t)obs[(size_t)(gbase + r) * IN_DIM + c];
  }

  // resident A operand: this wave's 16x512 H1 tile (128 VGPRs)
  v16bf a[16];
  const bf16_t* h1p = H1bf + (size_t)(gbase + wbase) * H_DIM;
#pragma unroll
  for (int kb = 0; kb < 16; ++kb) a[kb] = load_a16(h1p, H_DIM, kb, lane);

  // pipeline prologue: stage chunk 0
  stage_chunk(W2bf, chunk_base(0), &w2s[0][0][0], tid);
  ASYNC_WAIT();
  __syncthreads();

  int tf = 0;  // flat chunk index, advances with every consume below
  for (int layer = 0; layer < 3; ++layer) {
    const int oa = layer * 8192;
    const int ob = oa + 4096;
    const int d2 = (layer == 2) ? OUT_DIM : IN_DIM;

    // ---- A phase: y[b,r] = sum_d x[b,d] * (p[b, oa+d*16+r] + b2) ----------
    v8f yacc = {0.f, 0.f, 0.f, 0.f, 0.f, 0.f, 0.f, 0.f};
    for (int d = 0; d < IN_DIM; ++d) {
      if (tf + 1 < NCHUNK)
        stage_chunk(W2bf, chunk_base(tf + 1), &w2s[(tf + 1) & 1][0][0], tid);
      const bf16_t* bsrc = &w2s[tf & 1][0][0];
      v8f pc = {0.f, 0.f, 0.f, 0.f, 0.f, 0.f, 0.f, 0.f};
#pragma unroll
      for (int kb = 0; kb < 16; ++kb)
        pc = wmma_bf16(a[kb], load_b16(bsrc, H_DIM, kb, lane), pc);
      const float bias = b2[oa + d * 16 + n];
#pragma unroll
      for (int i = 0; i < 8; ++i) {
        const float xv = (float)xbuf[wbase + i + 8 * hi][d];
        yacc[i] += xv * (pc[i] + bias);
      }
      ASYNC_WAIT();
      __syncthreads();
      ++tf;
    }
#pragma unroll
    for (int i = 0; i < 8; ++i) ybuf[w][i + 8 * hi][n] = yacc[i];  // wave-local

    // ---- B phase: out[b,j] = sum_r y[b,r] * (p[b, ob+r*d2+j] + b2) --------
    const int nj = d2 >> 4;
    for (int j0 = 0; j0 < nj; ++j0) {
      v8f oacc = {0.f, 0.f, 0.f, 0.f, 0.f, 0.f, 0.f, 0.f};
      for (int r = 0; r < 16; ++r) {
        if (tf + 1 < NCHUNK)
          stage_chunk(W2bf, chunk_base(tf + 1), &w2s[(tf + 1) & 1][0][0], tid);
        const bf16_t* bsrc = &w2s[tf & 1][0][0];
        v8f pc = {0.f, 0.f, 0.f, 0.f, 0.f, 0.f, 0.f, 0.f};
#pragma unroll
        for (int kb = 0; kb < 16; ++kb)
          pc = wmma_bf16(a[kb], load_b16(bsrc, H_DIM, kb, lane), pc);
        const float bias = b2[ob + r * d2 + j0 * 16 + n];
#pragma unroll
        for (int i = 0; i < 8; ++i) {
          const float yv = ybuf[w][i + 8 * hi][r];
          oacc[i] += yv * (pc[i] + bias);
        }
        ASYNC_WAIT();
        __syncthreads();
        ++tf;
      }
      if (layer == 2) {  // final layer: no ReLU, fp32 output
#pragma unroll
        for (int i = 0; i < 8; ++i)
          out[(size_t)(gbase + wbase + i + 8 * hi) * OUT_DIM + j0 * 16 + n] =
              oacc[i];
      } else {  // ReLU, becomes next layer's x (wave-local xbuf region)
#pragma unroll
        for (int i = 0; i < 8; ++i)
          xbuf[wbase + i + 8 * hi][j0 * 16 + n] = (bf16_t)fmaxf(oacc[i], 0.f);
      }
    }
  }
}

// ---------------------------------------------------------------------------
extern "C" void kernel_launch(void* const* d_in, const int* in_sizes, int n_in,
                              void* d_out, int out_size, void* d_ws,
                              size_t ws_size, hipStream_t stream) {
  (void)in_sizes; (void)n_in; (void)out_size; (void)ws_size;

  const float* obs = (const float*)d_in[0];
  const float* par = (const float*)d_in[1];
  const float* W0  = (const float*)d_in[2];
  const float* b0  = (const float*)d_in[3];
  const float* W1  = (const float*)d_in[4];
  const float* b1  = (const float*)d_in[5];
  const float* W2  = (const float*)d_in[6];
  const float* b2  = (const float*)d_in[7];
  float* out = (float*)d_out;

  // bf16 staging buffers in workspace (~31.6 MB total)
  char* ws = (char*)d_ws;
  bf16_t* Pbf  = (bf16_t*)ws; ws += (size_t)B_TOT * P_DIM * 2;
  bf16_t* W0bf = (bf16_t*)ws; ws += (size_t)H_DIM * P_DIM * 2;
  bf16_t* W1bf = (bf16_t*)ws; ws += (size_t)H_DIM * H_DIM * 2;
  bf16_t* W2bf = (bf16_t*)ws; ws += (size_t)TOTP * H_DIM * 2;
  bf16_t* H1bf = (bf16_t*)ws; ws += (size_t)B_TOT * H_DIM * 2;

  auto cvt = [&](const float* s, bf16_t* d, int n) {
    int blocks = (n + 2047) / 2048;
    cvt_f32_bf16<<<blocks, 256, 0, stream>>>(s, d, n);
  };
  cvt(par, Pbf, B_TOT * P_DIM);
  cvt(W0, W0bf, H_DIM * P_DIM);
  cvt(W1, W1bf, H_DIM * H_DIM);
  cvt(W2, W2bf, TOTP * H_DIM);

  hyper_kernel<<<B_TOT / 32, 64, 0, stream>>>(Pbf, W0bf, b0, W1bf, b1, H1bf);
  target_kernel<<<B_TOT / 64, 128, 0, stream>>>(obs, H1bf, W2bf, b2, out);
}
